// coRNN_62835371540843
// MI455X (gfx1250) — matrix operand order repaired and checked by reading, not compiled, not run on hardware
//
#include <hip/hip_runtime.h>

// ---------------------------------------------------------------------------
// coRNN for MI455X (gfx1250, wave32):
//   1) u = x @ Wx_w^T + Wx_b   -> bf16 WMMA GEMM, M=16384 N=4096 K=1024
//      double-buffered LDS + async global->LDS (ASYNCcnt), wave tile 32x64
//      (2 A frags x 4 B frags -> 8 wmma, 24 B/lane/wmma of LDS reads).
//   2) sequential scan: h = tanh(u_t + circconv(h)), 1 WG / batch element,
//      hidden state resident in LDS, T-loop in-kernel, next-step u prefetch.
//   3) out = h @ Ro_w^T + Ro_b -> folded into scan epilogue (LDS reduction)
// ---------------------------------------------------------------------------

#define USE_ASYNC_LDS 1

typedef __bf16 bf16_t;
typedef __attribute__((ext_vector_type(4)))  __bf16 v4bf;
typedef __attribute__((ext_vector_type(8)))  __bf16 v8bf;
typedef __attribute__((ext_vector_type(16))) __bf16 v16bf;
typedef __attribute__((ext_vector_type(8)))  float  v8f;

// round-to-nearest-even fp32 -> bf16 (manual; avoids header dependence)
__device__ __forceinline__ bf16_t f2bf(float f) {
    union { float f; unsigned u; } c; c.f = f;
    unsigned r = c.u + 0x7FFFu + ((c.u >> 16) & 1u);
    unsigned short s = (unsigned short)(r >> 16);
    bf16_t b; __builtin_memcpy(&b, &s, 2); return b;
}

#if USE_ASYNC_LDS
// CDNA5 async DMA: global -> LDS, 16B per lane, tracked by ASYNCcnt.
// LDS dest address = low 32 bits of the flat shared pointer (HW uses
// addr[31:0] as the LDS byte address for the LDS aperture).
__device__ __forceinline__ void async_b128_to_lds(unsigned lds_addr,
                                                  const void* gaddr) {
    asm volatile("global_load_async_to_lds_b128 %0, %1, off"
                 :: "v"(lds_addr), "v"(gaddr) : "memory");
}
__device__ __forceinline__ void wait_asynccnt0() {
    asm volatile("s_wait_asynccnt 0x0" ::: "memory");
}
#endif

// ---------------------------------------------------------------------------
// fp32 -> bf16 conversion, 4 elements/thread
// ---------------------------------------------------------------------------
__global__ __launch_bounds__(256) void cvt_f32_bf16(const float* __restrict__ in,
                                                    bf16_t* __restrict__ out, int n) {
    int i = (blockIdx.x * 256 + threadIdx.x) * 4;
    if (i + 3 < n) {
        float4 v = *(const float4*)(in + i);
        v4bf o;
        o[0] = f2bf(v.x); o[1] = f2bf(v.y); o[2] = f2bf(v.z); o[3] = f2bf(v.w);
        *(v4bf*)(out + i) = o;
    }
}

// ---------------------------------------------------------------------------
// GEMM: C[M,N] = A[M,K] * Bw[N,K]^T + bias[N]    (A,Bw bf16, C fp32)
// 256 threads = 8 waves. Block tile 128x128, BK=32, double-buffered LDS.
// Waves arranged 4 (M) x 2 (N); each wave owns a 32x64 output tile:
//   2 A fragments x 4 B fragments -> 8 WMMAs with 4x A / 2x B reuse.
// ---------------------------------------------------------------------------
__global__ __launch_bounds__(256) void gemm_u_kernel(
    const bf16_t* __restrict__ A, const bf16_t* __restrict__ Bw,
    const float* __restrict__ bias, float* __restrict__ C,
    int M, int N, int K)
{
    constexpr int BM = 128, BN = 128, BK = 32, LDT = 40;  // 40 bf16 = 80B padded row
    __shared__ bf16_t As[2][BM * LDT];
    __shared__ bf16_t Bs[2][BN * LDT];

    const int tid    = threadIdx.x;
    const int wave   = tid >> 5;
    const int wave_m = wave & 3;        // 0..3 -> 32-row strip
    const int wave_n = wave >> 2;       // 0..1 -> 64-col strip
    const int lane   = tid & 31;
    const int l16    = lane & 15;
    const bool hih   = lane >= 16;
    const int m0 = blockIdx.y * BM;
    const int n0 = blockIdx.x * BN;

    v8f acc[2][4] = {};

    // Each thread moves 2 x 16B chunks per tile per matrix (128x32 bf16 tile).
    auto prefetch = [&](int buf, int k0) {
#pragma unroll
        for (int i = 0; i < 2; i++) {
            const int chunk = tid + i * 256;
            const int row = chunk >> 2;
            const int kc  = (chunk & 3) * 8;
            const bf16_t* ga = A  + (size_t)(m0 + row) * K + k0 + kc;
            const bf16_t* gb = Bw + (size_t)(n0 + row) * K + k0 + kc;
#if USE_ASYNC_LDS
            async_b128_to_lds((unsigned)(unsigned long long)&As[buf][row * LDT + kc], ga);
            async_b128_to_lds((unsigned)(unsigned long long)&Bs[buf][row * LDT + kc], gb);
#else
            *(v8bf*)&As[buf][row * LDT + kc] = *(const v8bf*)ga;
            *(v8bf*)&Bs[buf][row * LDT + kc] = *(const v8bf*)gb;
#endif
        }
    };

    prefetch(0, 0);
    const int KT = K / BK;

    for (int kt = 0; kt < KT; kt++) {
        const int p = kt & 1;
#if USE_ASYNC_LDS
        wait_asynccnt0();             // our async writes to buf p have landed
#endif
        __syncthreads();              // everyone's writes to buf p have landed
        if (kt + 1 < KT) prefetch(p ^ 1, (kt + 1) * BK);

        // A fragments (16x32): lanes 0-15 row=l16 K{0-7,16-23};
        //                      lanes 16-31 row=l16 K{8-15,24-31}
        const int akoff = hih ? 8 : 0;
        v16bf afrag[2];
#pragma unroll
        for (int mi = 0; mi < 2; mi++) {
            const int arow = wave_m * 32 + mi * 16 + l16;
            v8bf alo = *(const v8bf*)&As[p][arow * LDT + akoff];
            v8bf ahi = *(const v8bf*)&As[p][arow * LDT + akoff + 16];
            afrag[mi] = __builtin_shufflevector(alo, ahi,
                0,1,2,3,4,5,6,7,8,9,10,11,12,13,14,15);
        }

        // B fragments (32x16): lanes 0-15 col=l16 K{0-7,16-23}; lanes 16-31 K{8-15,24-31}
        const int bkoff = hih ? 8 : 0;
        v16bf bfrag[4];
#pragma unroll
        for (int nt = 0; nt < 4; nt++) {
            const int bcol = wave_n * 64 + nt * 16 + l16;
            v8bf blo = *(const v8bf*)&Bs[p][bcol * LDT + bkoff];
            v8bf bhi = *(const v8bf*)&Bs[p][bcol * LDT + bkoff + 16];
            bfrag[nt] = __builtin_shufflevector(blo, bhi,
                0,1,2,3,4,5,6,7,8,9,10,11,12,13,14,15);
        }

#pragma unroll
        for (int mi = 0; mi < 2; mi++)
#pragma unroll
            for (int nt = 0; nt < 4; nt++)
                acc[mi][nt] = __builtin_amdgcn_wmma_f32_16x16x32_bf16(
                    false, afrag[mi], false, bfrag[nt], (short)0,
                    acc[mi][nt], false, false);
        // No trailing barrier: next iteration's top barrier orders
        // this iteration's LDS reads against the following overwrites.
    }

    // Epilogue. C/D layout: VGPR r -> M=r (lanes 0-15) / M=r+8 (lanes 16-31), N=l16.
#pragma unroll
    for (int mi = 0; mi < 2; mi++) {
        const int mrow = m0 + wave_m * 32 + mi * 16 + (hih ? 8 : 0);
#pragma unroll
        for (int nt = 0; nt < 4; nt++) {
            const int col = n0 + wave_n * 64 + nt * 16 + l16;
            const float bv = bias[col];
#pragma unroll
            for (int r = 0; r < 8; r++)
                C[(size_t)(mrow + r) * N + col] = acc[mi][nt][r] + bv;
        }
    }
}

// ---------------------------------------------------------------------------
// Sequential scan + readout. One block per batch element; 1024 threads,
// thread j owns hidden column j of all 4 channels. Hidden state in LDS.
// conv: y[co,j] = b[co] + sum_ci sum_k W[co,ci,k] * h[ci,(j+k-1) mod 1024]
// ---------------------------------------------------------------------------
__global__ __launch_bounds__(1024) void scan_kernel(
    const float* __restrict__ u,     // [T, B, 4096]
    const float* __restrict__ Wy_w,  // [4,4,3]
    const float* __restrict__ Wy_b,  // [4]
    const float* __restrict__ Ro_w,  // [10, 4096]
    const float* __restrict__ Ro_b,  // [10]
    float* __restrict__ out,         // [B, 10]
    int T, int Bb, int n_out)
{
    constexpr int NCH = 4, NH = 1024, H = NCH * NH;
    __shared__ float h[H];
    __shared__ float wy[NCH * NCH * 3];
    __shared__ float wyb[NCH];
    __shared__ float red[NH];

    const int j = threadIdx.x;
    const int b = blockIdx.x;

#pragma unroll
    for (int c = 0; c < NCH; c++) h[c * NH + j] = 0.f;
    if (j < NCH * NCH * 3) wy[j] = Wy_w[j];
    if (j < NCH)           wyb[j] = Wy_b[j];
    __syncthreads();

    const int jm = (j + NH - 1) & (NH - 1);
    const int jp = (j + 1) & (NH - 1);

    for (int t = 0; t < T; t++) {
        const float* ut = u + ((size_t)t * Bb + b) * H;
        // Prefetch next step's u slice while computing this one (HBM latency
        // is the critical path of this sequential loop).
        if (t + 1 < T) {
            const float* un = u + ((size_t)(t + 1) * Bb + b) * H;
#pragma unroll
            for (int c = 0; c < NCH; c++)
                __builtin_prefetch(un + c * NH + j, 0, 1);
        }
        float nv[NCH];
#pragma unroll
        for (int co = 0; co < NCH; co++) {
            float acc = wyb[co];
#pragma unroll
            for (int ci = 0; ci < NCH; ci++) {
                const float* w = &wy[(co * NCH + ci) * 3];
                acc += w[0] * h[ci * NH + jm]
                     + w[1] * h[ci * NH + j]
                     + w[2] * h[ci * NH + jp];
            }
            nv[co] = tanhf(ut[co * NH + j] + acc);
        }
        __syncthreads();
#pragma unroll
        for (int co = 0; co < NCH; co++) h[co * NH + j] = nv[co];
        __syncthreads();
    }

    // Readout: out[b,o] = Ro_b[o] + sum_h hy[h] * Ro_w[o,h]
    for (int o = 0; o < n_out; o++) {
        float p = 0.f;
#pragma unroll
        for (int c = 0; c < NCH; c++)
            p += h[c * NH + j] * Ro_w[(size_t)o * H + c * NH + j];
        red[j] = p;
        __syncthreads();
        for (int s = NH / 2; s > 0; s >>= 1) {
            if (j < s) red[j] += red[j + s];
            __syncthreads();
        }
        if (j == 0) out[b * n_out + o] = red[0] + Ro_b[o];
        __syncthreads();
    }
}

// ---------------------------------------------------------------------------
extern "C" void kernel_launch(void* const* d_in, const int* in_sizes, int n_in,
                              void* d_out, int out_size, void* d_ws, size_t ws_size,
                              hipStream_t stream) {
    const float* x    = (const float*)d_in[0];  // [T,B,1024]
    const float* Wx_w = (const float*)d_in[1];  // [4096,1024]
    const float* Wx_b = (const float*)d_in[2];  // [4096]
    const float* Wy_w = (const float*)d_in[3];  // [4,4,3]
    const float* Wy_b = (const float*)d_in[4];  // [4]
    const float* Ro_w = (const float*)d_in[5];  // [10,4096]
    const float* Ro_b = (const float*)d_in[6];  // [10]
    float* out = (float*)d_out;

    const int n_inp = 1024, H = 4096, Bb = 64, n_out = 10;
    const int K = n_inp, N = H;
    const int M = in_sizes[0] / n_inp;  // T*B = 16384
    const int T = M / Bb;               // 256

    // Workspace layout (256B-aligned): bf16 x | bf16 Wx | fp32 u
    char* ws = (char*)d_ws;
    size_t off = 0;
    bf16_t* xb = (bf16_t*)(ws + off);
    off += ((size_t)M * K * sizeof(bf16_t) + 255) & ~(size_t)255;
    bf16_t* wb = (bf16_t*)(ws + off);
    off += ((size_t)N * K * sizeof(bf16_t) + 255) & ~(size_t)255;
    float* u = (float*)(ws + off);      // M*N fp32 = 256 MB

    // 1) fp32 -> bf16 conversions
    cvt_f32_bf16<<<(M * K) / 1024, 256, 0, stream>>>(x, xb, M * K);
    cvt_f32_bf16<<<(N * K) / 1024, 256, 0, stream>>>(Wx_w, wb, N * K);

    // 2) u = x @ Wx_w^T + Wx_b  (WMMA bf16, fp32 accumulate)
    dim3 ggrid(N / 128, M / 128);
    gemm_u_kernel<<<ggrid, 256, 0, stream>>>(xb, wb, Wx_b, u, M, N, K);

    // 3) scan over T + readout
    scan_kernel<<<Bb, 1024, 0, stream>>>(u, Wy_w, Wy_b, Ro_w, Ro_b, out,
                                         T, Bb, n_out);
}
